// HidsModel_5153960755713
// MI455X (gfx1250) — matrix operand order repaired and checked by reading, not compiled, not run on hardware
//
#include <hip/hip_runtime.h>
#include <hip/hip_bf16.h>

// ---------------------------------------------------------------------------
// 9-layer LSTM (T=2048, B=32, IN=128, H=256, OUT=2) for gfx1250 / MI455X.
// Per layer: one persistent 1024-thread workgroup scans t=0..T-1.
// Each step: g = [h_{t-1} ; x_t] (bf16, LDS) x W_packed (bf16, L2-resident)
// via v_wmma_f32_16x16x32_bf16; gates fused via wave shuffles; cell state c
// kept in LDS (frees 32 VGPRs -> no scratch spills in the hot loop);
// x_{t+1} staged into a double-buffered LDS slot with
// GLOBAL_LOAD_ASYNC_TO_LDS_B128 overlapped with the GEMM (ASYNCcnt).
// ---------------------------------------------------------------------------

#define TT      2048
#define BB      32
#define IN_DIM  128
#define HID     256
#define G4H     1024      // 4*H
#define NL      9
#define OUT_DIM 2
#define HPAD    264       // LDS row stride (ushorts): 528B -> conflict-free
#define CPAD    260       // LDS row stride (floats) for cell state

typedef __bf16        v16bf __attribute__((ext_vector_type(16)));
typedef float         v8f   __attribute__((ext_vector_type(8)));
typedef unsigned int  v8u   __attribute__((ext_vector_type(8)));

union Frag {
    v8u          u8;
    v16bf        b;
    unsigned int u[8];
};

__device__ __forceinline__ unsigned short f2bf(float f) {
    unsigned u = __float_as_uint(f);
    unsigned r = u + 0x7FFFu + ((u >> 16) & 1u);   // round-to-nearest-even
    return (unsigned short)(r >> 16);
}
__device__ __forceinline__ float bf2f(unsigned short u) {
    return __uint_as_float(((unsigned)u) << 16);
}
__device__ __forceinline__ float sigf(float x) {
    return 1.0f / (1.0f + __expf(-x));
}
__device__ __forceinline__ void wait_asynccnt0() {
#if __has_builtin(__builtin_amdgcn_s_wait_asynccnt)
    __builtin_amdgcn_s_wait_asynccnt(0);
#else
    asm volatile("s_wait_asynccnt 0x0" ::: "memory");
#endif
}

// ---------------------------------------------------------------------------
// Pack [Whh ; Wih] (stacked along K) into per-lane WMMA B-fragment order,
// with gate-interleaved columns: packed col n' = unit*4 + gate
// (original col = gate*256 + unit).  Layout:
//   out[((nt*ktiles + kt)*32 + lane)*8 + p]  = bf16 pair (K=k0, K=k0+1)
//   lanes 0-15: K = kt*32 + 0..15 ; lanes 16-31: K = kt*32 + 16..31
// ---------------------------------------------------------------------------
__global__ void pack_weights_kernel(const float* __restrict__ Whh,   // [4H][H]
                                    const float* __restrict__ Wih,   // [4H][Din]
                                    int Din, int ktiles,
                                    unsigned int* __restrict__ out) {
    int idx   = blockIdx.x * blockDim.x + threadIdx.x;
    int total = 64 * ktiles * 32 * 8;
    if (idx >= total) return;
    int p    = idx & 7;
    int rest = idx >> 3;
    int lane = rest & 31;
    rest >>= 5;
    int kt = rest % ktiles;
    int nt = rest / ktiles;

    int nprime = nt * 16 + (lane & 15);
    int unit   = nprime >> 2;
    int gate   = nprime & 3;
    int ncol   = gate * HID + unit;          // original 4H row index

    int khalf = (lane & 16) ? 16 : 0;
    int k0    = kt * 32 + khalf + 2 * p;

    float w0, w1;
    if (k0 < HID) {                          // recurrent part
        w0 = Whh[(size_t)ncol * HID + k0];
        w1 = Whh[(size_t)ncol * HID + k0 + 1];
    } else {                                 // input-projection part
        int kk = k0 - HID;
        w0 = Wih[(size_t)ncol * Din + kk];
        w1 = Wih[(size_t)ncol * Din + kk + 1];
    }
    out[idx] = (unsigned)f2bf(w0) | ((unsigned)f2bf(w1) << 16);
}

// ---------------------------------------------------------------------------
// A-fragment gather from an LDS region (bf16 row-major [32][HPAD]).
// A k order used here (contiguous, same as packed B): lanes 0-15 -> K 0..15, lanes 16-31 -> K 16..31.
// ktl is the k-tile index local to the region (each region covers K=256 max).
// ---------------------------------------------------------------------------
__device__ __forceinline__ void loadA(Frag& f, const unsigned short* region,
                                      int mt, int ktl, int lid) {
    int m = mt * 16 + (lid & 15);
    const unsigned* row = reinterpret_cast<const unsigned*>(region + (size_t)m * HPAD);
    int kbase = ktl * 32 + ((lid & 16) ? 16 : 0);
#pragma unroll
    for (int p = 0; p < 8; ++p) {
        int k = kbase + 2 * p;
        f.u[p] = row[k >> 1];
    }
}

__device__ __forceinline__ void loadB(Frag& f, const v8u* __restrict__ pB,
                                      int nt, int kt, int ktiles, int lid) {
    f.u8 = pB[(size_t)(nt * ktiles + kt) * 32 + lid];
}

// ---------------------------------------------------------------------------
// Gate fusion: packed col n' = unit*4 + gate, so the 4 gate pre-activations
// for a unit sit in 4 adjacent lanes of each accumulator VGPR.  Cell state
// lives in LDS; only the quad's writer lane does the c read-modify-write.
// ---------------------------------------------------------------------------
__device__ __forceinline__ void gates_update(const v8f& acc,
                                             int mt, int nt, int lid, int t,
                                             unsigned short* sH, float* sC,
                                             unsigned short* __restrict__ h_out) {
    int  base   = lid & ~3;
    bool writer = (lid & 3) == 0;
    int  unit   = nt * 4 + ((lid & 15) >> 2);
#pragma unroll
    for (int r = 0; r < 8; ++r) {
        float a  = acc[r];
        float gi = __shfl(a, base + 0);
        float gf = __shfl(a, base + 1);
        float gg = __shfl(a, base + 2);
        float go = __shfl(a, base + 3);
        if (writer) {
            int m  = (lid < 16) ? r : r + 8;
            int bb = mt * 16 + m;
            float cc = sC[(size_t)bb * CPAD + unit];
            cc = sigf(gf) * cc + sigf(gi) * tanhf(gg);
            sC[(size_t)bb * CPAD + unit] = cc;
            float hh = sigf(go) * tanhf(cc);
            unsigned short hb = f2bf(hh);
            sH[(size_t)bb * HPAD + unit] = hb;
            h_out[((size_t)t * BB + bb) * HID + unit] = hb;
        }
    }
}

// Synchronous staging (layer 0 f32 input, and the t=0 prologue for all layers).
__device__ __forceinline__ void stage_x_sync(unsigned short* dst,
                                             const float* __restrict__ xf,
                                             const unsigned short* __restrict__ xb,
                                             int t, int tid) {
    if (xf) {   // layer 0: f32 input, Din = 128
        for (int i = tid; i < BB * IN_DIM; i += 1024) {
            int bb = i >> 7, k = i & 127;
            dst[(size_t)bb * HPAD + k] =
                f2bf(xf[((size_t)t * BB + bb) * IN_DIM + k]);
        }
    } else {    // layers 1..8: bf16 hidden sequence, Din = 256
        for (int i = tid; i < BB * HID; i += 1024) {
            int bb = i >> 8, k = i & 255;
            dst[(size_t)bb * HPAD + k] =
                xb[((size_t)t * BB + bb) * HID + k];
        }
    }
}

// Async staging: one GLOBAL_LOAD_ASYNC_TO_LDS_B128 per thread covers the
// whole 32x256 bf16 tile (1024 threads x 16B).  Tracked by ASYNCcnt.
__device__ __forceinline__ void stage_x_async(unsigned short* dst,
                                              const unsigned short* __restrict__ xb,
                                              int t, int tid) {
    int bb    = tid >> 5;      // row 0..31
    int chunk = tid & 31;      // 16B chunk within row
    unsigned lds_off = (unsigned)(uintptr_t)(dst + (size_t)bb * HPAD + chunk * 8);
    const unsigned short* gp = xb + ((size_t)t * BB + bb) * HID + chunk * 8;
    asm volatile("global_load_async_to_lds_b128 %0, %1, off"
                 :: "v"(lds_off), "v"(gp) : "memory");
}

// ---------------------------------------------------------------------------
// Persistent scan over T for one LSTM layer.  1 block x 1024 threads (32
// waves).  Wave w owns C-tiles (mt in {0,1}) x (nt in {2w, 2w+1}).
// ---------------------------------------------------------------------------
__global__ void __launch_bounds__(1024)
lstm_scan_kernel(const unsigned int* __restrict__ pW,   // packed bf16 weights
                 const float* __restrict__ bias,        // [4H]
                 const float* __restrict__ x_f32,       // layer 0 input or null
                 const unsigned short* __restrict__ x_bf,// layer>0 input or null
                 unsigned short* __restrict__ h_out,    // [T][B][H] bf16
                 int ktiles) {
    __shared__ __align__(16) unsigned short sH[BB * HPAD];      // h_{t-1} bf16
    __shared__ __align__(16) unsigned short sX[2][BB * HPAD];   // x double buffer
    __shared__ __align__(16) float          sC[BB * CPAD];      // cell state f32

    const int tid  = threadIdx.x;
    const int lid  = tid & 31;
    const int wave = tid >> 5;
    const int nt0  = wave * 2;
    const int nt1  = wave * 2 + 1;
    const v8u* pB  = reinterpret_cast<const v8u*>(pW);

    // Per-lane bias (constant over t), gate-interleaved column mapping.
    int np0 = nt0 * 16 + (lid & 15);
    int np1 = nt1 * 16 + (lid & 15);
    const float bv0 = bias[(np0 & 3) * HID + (np0 >> 2)];
    const float bv1 = bias[(np1 & 3) * HID + (np1 >> 2)];

    // h_0 = 0 ; c_0 = 0 ; stage x_0 into slot 0.
    for (int i = tid; i < BB * HID; i += 1024) {
        sH[(size_t)(i >> 8) * HPAD + (i & 255)] = 0;
        sC[(size_t)(i >> 8) * CPAD + (i & 255)] = 0.0f;
    }
    stage_x_sync(&sX[0][0], x_f32, x_bf, 0, tid);
    __syncthreads();

    for (int t = 0; t < TT; ++t) {
        unsigned short* xcur = &sX[t & 1][0];
        unsigned short* xnxt = &sX[(t & 1) ^ 1][0];

        // Overlap next step's input copy with this step's GEMM (ASYNCcnt).
        if (x_bf && (t + 1 < TT)) stage_x_async(xnxt, x_bf, t + 1, tid);

        v8f a00, a01, a10, a11;
#pragma unroll
        for (int j = 0; j < 8; ++j) { a00[j] = bv0; a01[j] = bv1; a10[j] = bv0; a11[j] = bv1; }

        // Recurrent half: K = 0..255 from sH (k-tiles 0..7).
#pragma unroll 4
        for (int kt = 0; kt < 8; ++kt) {
            Frag fa0, fa1, fb0, fb1;
            loadA(fa0, sH, 0, kt, lid);
            loadA(fa1, sH, 1, kt, lid);
            loadB(fb0, pB, nt0, kt, ktiles, lid);
            loadB(fb1, pB, nt1, kt, ktiles, lid);
            a00 = __builtin_amdgcn_wmma_f32_16x16x32_bf16(false, fa0.b, false, fb0.b,
                                                          (short)0, a00, false, false);
            a01 = __builtin_amdgcn_wmma_f32_16x16x32_bf16(false, fa0.b, false, fb1.b,
                                                          (short)0, a01, false, false);
            a10 = __builtin_amdgcn_wmma_f32_16x16x32_bf16(false, fa1.b, false, fb0.b,
                                                          (short)0, a10, false, false);
            a11 = __builtin_amdgcn_wmma_f32_16x16x32_bf16(false, fa1.b, false, fb1.b,
                                                          (short)0, a11, false, false);
        }
        // Input half: K = 256.. from current x slot (k-tiles 8..ktiles-1).
        for (int kt = 8; kt < ktiles; ++kt) {
            Frag fa0, fa1, fb0, fb1;
            loadA(fa0, xcur, 0, kt - 8, lid);
            loadA(fa1, xcur, 1, kt - 8, lid);
            loadB(fb0, pB, nt0, kt, ktiles, lid);
            loadB(fb1, pB, nt1, kt, ktiles, lid);
            a00 = __builtin_amdgcn_wmma_f32_16x16x32_bf16(false, fa0.b, false, fb0.b,
                                                          (short)0, a00, false, false);
            a01 = __builtin_amdgcn_wmma_f32_16x16x32_bf16(false, fa0.b, false, fb1.b,
                                                          (short)0, a01, false, false);
            a10 = __builtin_amdgcn_wmma_f32_16x16x32_bf16(false, fa1.b, false, fb0.b,
                                                          (short)0, a10, false, false);
            a11 = __builtin_amdgcn_wmma_f32_16x16x32_bf16(false, fa1.b, false, fb1.b,
                                                          (short)0, a11, false, false);
        }
        __syncthreads();   // all LDS reads of h_{t-1} / x_t complete

        gates_update(a00, 0, nt0, lid, t, sH, sC, h_out);
        gates_update(a01, 0, nt1, lid, t, sH, sC, h_out);
        gates_update(a10, 1, nt0, lid, t, sH, sC, h_out);
        gates_update(a11, 1, nt1, lid, t, sH, sC, h_out);
        if (x_f32 && (t + 1 < TT)) stage_x_sync(xnxt, x_f32, nullptr, t + 1, tid);
        wait_asynccnt0();  // this wave's async LDS writes landed
        __syncthreads();   // h_t / x_{t+1} visible to all waves
    }
}

// ---------------------------------------------------------------------------
// Output head: out[t,b,:] = sigmoid(h[t,b,:] @ Wo^T + bo), OUT=2.
// ---------------------------------------------------------------------------
__global__ void head_kernel(const unsigned short* __restrict__ h,  // [T][B][H]
                            const float* __restrict__ Wo,          // [2][H]
                            const float* __restrict__ bo,          // [2]
                            float* __restrict__ out) {             // [T][B][2]
    int i = blockIdx.x * blockDim.x + threadIdx.x;   // over T*B
    if (i >= TT * BB) return;
    const unsigned short* hr = h + (size_t)i * HID;
    float a0 = bo[0], a1 = bo[1];
#pragma unroll 4
    for (int k = 0; k < HID; ++k) {
        float hv = bf2f(hr[k]);
        a0 += hv * Wo[k];
        a1 += hv * Wo[HID + k];
    }
    out[(size_t)i * 2 + 0] = sigf(a0);
    out[(size_t)i * 2 + 1] = sigf(a1);
}

// ---------------------------------------------------------------------------
extern "C" void kernel_launch(void* const* d_in, const int* in_sizes, int n_in,
                              void* d_out, int out_size, void* d_ws, size_t ws_size,
                              hipStream_t stream) {
    const float* x        = (const float*)d_in[0];   // [T][B][IN]
    const float* Wih0     = (const float*)d_in[1];   // [4H][IN]
    const float* Wih_rest = (const float*)d_in[2];   // [L-1][4H][H]
    const float* Whh      = (const float*)d_in[3];   // [L][4H][H]
    const float* b        = (const float*)d_in[4];   // [L][4H]
    const float* Wo       = (const float*)d_in[5];   // [2][H]
    const float* bo       = (const float*)d_in[6];   // [2]
    float* out            = (float*)d_out;

    char* ws = (char*)d_ws;
    unsigned int* pw[NL];
    size_t off = 0;
    for (int l = 0; l < NL; ++l) {
        int ktiles = (l == 0) ? ((HID + IN_DIM) / 32) : ((HID + HID) / 32);
        pw[l] = (unsigned int*)(ws + off);
        off += (size_t)64 * ktiles * 32 * 8 * sizeof(unsigned int);
    }
    const size_t hbytes = (size_t)TT * BB * HID * sizeof(unsigned short);
    unsigned short* hA = (unsigned short*)(ws + off);
    unsigned short* hB = (unsigned short*)(ws + off + hbytes);

    // 1) pack weights (bf16, WMMA-B fragment order, gate-interleaved cols)
    for (int l = 0; l < NL; ++l) {
        const float* whh_l = Whh + (size_t)l * G4H * HID;
        const float* wih_l = (l == 0) ? Wih0 : (Wih_rest + (size_t)(l - 1) * G4H * HID);
        int Din    = (l == 0) ? IN_DIM : HID;
        int ktiles = (HID + Din) / 32;
        int total  = 64 * ktiles * 32 * 8;
        pack_weights_kernel<<<(total + 255) / 256, 256, 0, stream>>>(
            whh_l, wih_l, Din, ktiles, pw[l]);
    }

    // 2) sequential per-layer scans (ping-pong hidden sequence buffers)
    unsigned short* cur_out = hA;
    unsigned short* cur_in  = hB;
    for (int l = 0; l < NL; ++l) {
        int ktiles = (l == 0) ? ((HID + IN_DIM) / 32) : ((HID + HID) / 32);
        lstm_scan_kernel<<<1, 1024, 0, stream>>>(
            pw[l], b + (size_t)l * G4H,
            (l == 0) ? x : nullptr,
            (l == 0) ? nullptr : cur_in,
            cur_out, ktiles);
        unsigned short* tmp = cur_in; cur_in = cur_out; cur_out = tmp;
    }
    // after the loop, the last layer's output is in cur_in

    // 3) output head
    head_kernel<<<(TT * BB + 255) / 256, 256, 0, stream>>>(cur_in, Wo, bo, out);
}
